// MultiHeadAttention_91079076479782
// MI455X (gfx1250) — matrix operand order repaired and checked by reading, not compiled, not run on hardware
//
#include <hip/hip_runtime.h>

// ---------------------------------------------------------------------------
// MHA forward for B=2, S=2048, D=1024, H=16, DH=64 on gfx1250 (wave32, WMMA).
// All matmuls use v_wmma_f32_16x16x32_bf16 (bf16 in, f32 accumulate).
// GEMM: double-buffered async global->LDS staging (ASYNCcnt), packed b128
// fragment loads per the ISA VGPR layout tables. Weights consumed in their
// native (e,d) = NxK layout (n-major, contiguous k) -> no transpose needed.
// ---------------------------------------------------------------------------

typedef __attribute__((ext_vector_type(16))) __bf16 v16bf;
typedef __attribute__((ext_vector_type(8)))  float  v8f;
typedef __attribute__((ext_vector_type(4)))  int    v4i;

typedef v4i __attribute__((address_space(1)))* glob_v4i;
typedef v4i __attribute__((address_space(3)))* lds_v4i;

union FragAB {
    v16bf    v;
    unsigned u[8];
    uint4    q[2];
};

#if defined(__has_builtin)
#if __has_builtin(__builtin_amdgcn_global_load_async_to_lds_b128)
#define HAVE_ASYNC_LDS 1
#endif
#endif
#ifndef HAVE_ASYNC_LDS
#define HAVE_ASYNC_LDS 0
#endif

static __device__ __forceinline__ unsigned short f2bf(float x) {
    union { float f; unsigned u; } v; v.f = x;
    unsigned r = v.u + 0x7FFFu + ((v.u >> 16) & 1u);   // round-to-nearest-even
    return (unsigned short)(r >> 16);
}
static __device__ __forceinline__ float bf2f(unsigned short s) {
    union { unsigned u; float f; } v; v.u = ((unsigned)s) << 16;
    return v.f;
}

static __device__ __forceinline__ v8f wmma_bf16(const FragAB& a, const FragAB& b, v8f c) {
    return __builtin_amdgcn_wmma_f32_16x16x32_bf16(
        false, a.v, false, b.v, (short)0, c, false, false);
}

// 16B global -> LDS copy: async TDM-style path when available.
static __device__ __forceinline__ void copy_b128(const unsigned short* g, unsigned short* d) {
#if HAVE_ASYNC_LDS
    __builtin_amdgcn_global_load_async_to_lds_b128((glob_v4i)(g), (lds_v4i)(d), 0, 0);
#else
    *(uint4*)d = *(const uint4*)g;
#endif
}
static __device__ __forceinline__ void wait_async_lds() {
#if HAVE_ASYNC_LDS
#if __has_builtin(__builtin_amdgcn_s_wait_asynccnt)
    __builtin_amdgcn_s_wait_asynccnt(0);
#else
    asm volatile("s_wait_asynccnt 0x0" ::: "memory");
#endif
#endif
}

// A-fragment (16x32): lane holds row m = lane&15; half hh = lane>>4 owns
// k in [8hh..8hh+7] (VGPR 0-3) and [16+8hh..16+8hh+7] (VGPR 4-7) -> 2x b128.
static __device__ __forceinline__ void loadA(FragAB& f, const unsigned short* row, int hh) {
    f.q[0] = *(const uint4*)(row + 8 * hh);
    f.q[1] = *(const uint4*)(row + 16 + 8 * hh);
}
// B-fragment (32x16): lane holds col n = lane&15; half hh owns
// k in [8hh..8hh+7] and [16+8hh..16+8hh+7] (same as A) -> 2x b128.
static __device__ __forceinline__ void loadB(FragAB& f, const unsigned short* col, int hh) {
    f.q[0] = *(const uint4*)(col + 8 * hh);
    f.q[1] = *(const uint4*)(col + 16 + 8 * hh);
}

// ---------------------------------------------------------------------------
// fp32 -> bf16 elementwise
// ---------------------------------------------------------------------------
__global__ void cvt_f32_bf16(const float* __restrict__ src,
                             unsigned short* __restrict__ dst, int n) {
    int i = blockIdx.x * blockDim.x + threadIdx.x;
    if (i < n) dst[i] = f2bf(src[i]);
}

// ---------------------------------------------------------------------------
// GEMM: out[m,n] = sum_k A[m,k]*B[n,k].  A: MxK bf16 row-major,
// Bn: NxK bf16 row-major (weights in native (e,d) layout).
// WG = 256 threads (8 waves), tile 128x64, Kstep 32, double-buffered LDS
// staged entirely with async b128 global->LDS loads.
// mode 0: fp32 row-major out            (final O projection)
// mode 1: bf16 out -> (B,H,S,64)        (Q,K projections)
// mode 2: bf16 out -> (B,H,64,S) V^T    (V projection, PV-friendly)
// ---------------------------------------------------------------------------
__global__ __launch_bounds__(256) void gemm_bf16(
    const unsigned short* __restrict__ Am,
    const unsigned short* __restrict__ Bn,
    float* __restrict__ outF,
    unsigned short* __restrict__ outH,
    int M, int N, int K, int mode)
{
    __shared__ unsigned short sA[2][128 * 32];   // m-major tiles
    __shared__ unsigned short sB[2][64 * 32];    // n-major tiles

    const int tid  = threadIdx.x;
    const int wv   = tid >> 5;
    const int lane = tid & 31;
    const int n    = lane & 15;
    const int hh   = lane >> 4;
    const int n0   = blockIdx.x * 64;
    const int m0   = blockIdx.y * 128;

    // per-thread staging addresses (row-contiguous 16B chunks)
    const int arow = tid >> 1, aq = tid & 1;     // A: 128 rows x (2 x 16B)
    const int brow = tid >> 2, bq = tid & 3;     // B:  64 rows x (4 x  8B->16B quads)

    v8f acc[4] = {};

    auto stage = [&](int buf, int kk) {
        const unsigned short* ga = Am + (size_t)(m0 + arow) * K + kk + aq * 16;
        unsigned short* da = sA[buf] + arow * 32 + aq * 16;
        copy_b128(ga, da);
        copy_b128(ga + 8, da + 8);
        const unsigned short* gb = Bn + (size_t)(n0 + brow) * K + kk + bq * 8;
        unsigned short* db = sB[buf] + brow * 32 + bq * 8;
        copy_b128(gb, db);
    };

    stage(0, 0);
    int buf = 0;
    for (int kk = 0; kk < K; kk += 32) {
        wait_async_lds();
        __syncthreads();
        if (kk + 32 < K) stage(buf ^ 1, kk + 32);

        FragAB a;
        loadA(a, sA[buf] + (wv * 16 + n) * 32, hh);
#pragma unroll
        for (int nt = 0; nt < 4; nt++) {
            FragAB b;
            loadB(b, sB[buf] + (nt * 16 + n) * 32, hh);
            acc[nt] = wmma_bf16(a, b, acc[nt]);
        }
        buf ^= 1;
    }

#pragma unroll
    for (int nt = 0; nt < 4; nt++) {
#pragma unroll
        for (int r = 0; r < 8; r++) {
            int gm = m0 + wv * 16 + r + 8 * hh;
            int gn = n0 + nt * 16 + n;
            float val = acc[nt][r];
            if (mode == 0) {
                outF[(size_t)gm * N + gn] = val;
            } else {
                int bb = gm >> 11, s = gm & 2047;       // gm = b*2048 + s
                int hd = gn >> 6,  dh = gn & 63;        // gn = h*64 + dh
                if (mode == 1)
                    outH[(((size_t)bb * 16 + hd) * 2048 + s) * 64 + dh] = f2bf(val);
                else // mode 2: V^T per head (B,H,64,S)
                    outH[(((size_t)bb * 16 + hd) * 64 + dh) * 2048 + s] = f2bf(val);
            }
        }
    }
}

// ---------------------------------------------------------------------------
// RMS-norm (over DH=64) + partial RoPE, in place on (B,H,S,64) bf16.
// ---------------------------------------------------------------------------
__global__ __launch_bounds__(256) void rmsnorm_rope(unsigned short* __restrict__ t,
                                                    int S, int nrows) {
    int wv = threadIdx.x >> 5, lane = threadIdx.x & 31;
    int row = blockIdx.x * 8 + wv;
    if (row >= nrows) return;
    int s = row % S;
    unsigned short* p = t + (size_t)row * 64;

    float x1 = bf2f(p[lane]);
    float x2 = bf2f(p[lane + 32]);
    float ss = x1 * x1 + x2 * x2;
#pragma unroll
    for (int off = 16; off; off >>= 1) ss += __shfl_xor(ss, off);
    float sc = rsqrtf(ss * (1.0f / 64.0f) + 1e-6f);
    x1 *= sc;
    x2 *= sc;

    float c = 1.0f, sn = 0.0f;
    if (lane < 16) {
        float fr = __powf(1024.0f, -(float)lane * (1.0f / 15.0f));
        __sincosf((float)s * fr, &sn, &c);
    }
    float y1 = x1 * c + x2 * sn;
    float y2 = x2 * c - x1 * sn;
    p[lane]      = f2bf(y1);
    p[lane + 32] = f2bf(y2);
}

// ---------------------------------------------------------------------------
// Flash attention, causal. grid (S/64, H, B), 128 threads = 4 waves.
// Each wave owns 16 query rows; K/V consumed in 32-column blocks.
// Q/K in (B,H,S,64); V pre-transposed (B,H,64,S) so all fragments are b128.
// ---------------------------------------------------------------------------
__global__ __launch_bounds__(128) void flash_attn(
    const unsigned short* __restrict__ Q,
    const unsigned short* __restrict__ Kc,
    const unsigned short* __restrict__ Vt,
    unsigned short* __restrict__ AO)
{
    const int Sn = 2048, Hn = 16, Dm = 1024;
    const int wv = threadIdx.x >> 5, lane = threadIdx.x & 31;
    const int n = lane & 15, hh = lane >> 4;
    const int qblk = blockIdx.x, h = blockIdx.y, b = blockIdx.z;
    const int s_base = qblk * 64 + wv * 16;

    const size_t headoff = (size_t)(b * Hn + h) * Sn * 64;
    const unsigned short* qh = Q + headoff;
    const unsigned short* kh = Kc + headoff;
    const unsigned short* vh = Vt + headoff;   // (64, S) per head

    __shared__ unsigned short sP[4][512];      // per-wave 16x32 bf16 P tile
    unsigned short* myP = sP[wv];

    FragAB aq0, aq1;
    {
        const unsigned short* qrow = qh + (size_t)(s_base + n) * 64;
        loadA(aq0, qrow, hh);
        loadA(aq1, qrow + 32, hh);
    }

    float mrow[8], lrow[8];
    v8f o[4] = {};
#pragma unroll
    for (int r = 0; r < 8; r++) { mrow[r] = -3.0e38f; lrow[r] = 0.0f; }

    for (int t0 = 0; t0 < s_base + 16; t0 += 32) {
        // ----- scores: S = Q * K^T (two 16x16 n-tiles) -----
        v8f sc0 = {}, sc1 = {};
        {
            FragAB bk;
            const unsigned short* krow0 = kh + (size_t)(t0 + n) * 64;
            const unsigned short* krow1 = kh + (size_t)(t0 + 16 + n) * 64;
            loadB(bk, krow0, hh);       sc0 = wmma_bf16(aq0, bk, sc0);
            loadB(bk, krow0 + 32, hh);  sc0 = wmma_bf16(aq1, bk, sc0);
            loadB(bk, krow1, hh);       sc1 = wmma_bf16(aq0, bk, sc1);
            loadB(bk, krow1 + 32, hh);  sc1 = wmma_bf16(aq1, bk, sc1);
        }

        // ----- online softmax update (per-row stats across 16 lanes) -----
        float p0[8], p1[8], alpha[8];
#pragma unroll
        for (int r = 0; r < 8; r++) {
            int srow = s_base + r + 8 * hh;
            float v0 = sc0[r] * 0.125f;
            float v1 = sc1[r] * 0.125f;
            if (t0 + n > srow)      v0 = -3.0e38f;
            if (t0 + 16 + n > srow) v1 = -3.0e38f;
            float rm = fmaxf(v0, v1);
            rm = fmaxf(rm, __shfl_xor(rm, 1));
            rm = fmaxf(rm, __shfl_xor(rm, 2));
            rm = fmaxf(rm, __shfl_xor(rm, 4));
            rm = fmaxf(rm, __shfl_xor(rm, 8));
            float m2 = fmaxf(mrow[r], rm);
            float a  = __expf(mrow[r] - m2);
            float e0 = __expf(v0 - m2);
            float e1 = __expf(v1 - m2);
            float rs = e0 + e1;
            rs += __shfl_xor(rs, 1);
            rs += __shfl_xor(rs, 2);
            rs += __shfl_xor(rs, 4);
            rs += __shfl_xor(rs, 8);
            lrow[r] = lrow[r] * a + rs;
            mrow[r] = m2;
            alpha[r] = a;
            p0[r] = e0;
            p1[r] = e1;
        }
#pragma unroll
        for (int dt = 0; dt < 4; dt++)
#pragma unroll
            for (int r = 0; r < 8; r++) o[dt][r] *= alpha[r];

        // ----- P (C/D layout) -> LDS -> A-fragment layout -----
#pragma unroll
        for (int r = 0; r < 8; r++) {
            myP[(r + 8 * hh) * 32 + n]      = f2bf(p0[r]);
            myP[(r + 8 * hh) * 32 + 16 + n] = f2bf(p1[r]);
        }
        FragAB ap;
        loadA(ap, myP + n * 32, hh);

        // ----- O += P * V  (V^T rows are t-contiguous -> packed B frags) ---
#pragma unroll
        for (int dt = 0; dt < 4; dt++) {
            FragAB bv;
            loadB(bv, vh + (size_t)(dt * 16 + n) * Sn + t0, hh);
            o[dt] = wmma_bf16(ap, bv, o[dt]);
        }
    }

    // ----- normalize and write to (B,S,D) bf16 -----
#pragma unroll
    for (int r = 0; r < 8; r++) {
        float inv = 1.0f / lrow[r];
        int srow = s_base + r + 8 * hh;
        size_t base = ((size_t)b * Sn + srow) * Dm + h * 64;
#pragma unroll
        for (int dt = 0; dt < 4; dt++)
            AO[base + dt * 16 + n] = f2bf(o[dt][r] * inv);
    }
}

// ---------------------------------------------------------------------------
// launch
// ---------------------------------------------------------------------------
extern "C" void kernel_launch(void* const* d_in, const int* in_sizes, int n_in,
                              void* d_out, int out_size, void* d_ws, size_t ws_size,
                              hipStream_t stream)
{
    (void)in_sizes; (void)n_in; (void)out_size; (void)ws_size;
    const float* x   = (const float*)d_in[0];
    const float* w_q = (const float*)d_in[1];
    const float* w_k = (const float*)d_in[2];
    const float* w_v = (const float*)d_in[3];
    const float* w_o = (const float*)d_in[4];
    // d_in[5] = causal mask (known tril, not needed on device)

    char* ws = (char*)d_ws;                                    // needs 48 MB
    unsigned short* xb  = (unsigned short*)(ws);               //  8 MB (B*S,D)
    unsigned short* wqb = (unsigned short*)(ws + (8u  << 20)); //  2 MB each, (e,d)=NxK
    unsigned short* wkb = (unsigned short*)(ws + (10u << 20));
    unsigned short* wvb = (unsigned short*)(ws + (12u << 20));
    unsigned short* wob = (unsigned short*)(ws + (14u << 20));
    unsigned short* qB  = (unsigned short*)(ws + (16u << 20)); //  8 MB (B,H,S,64)
    unsigned short* kB  = (unsigned short*)(ws + (24u << 20)); //  8 MB (B,H,S,64)
    unsigned short* vT  = (unsigned short*)(ws + (32u << 20)); //  8 MB (B,H,64,S)
    unsigned short* aoB = (unsigned short*)(ws + (40u << 20)); //  8 MB (B,S,D)

    const int M = 4096, N = 1024, K = 1024;

    cvt_f32_bf16<<<16384, 256, 0, stream>>>(x, xb, 4194304);
    cvt_f32_bf16<<<4096, 256, 0, stream>>>(w_q, wqb, 1048576);
    cvt_f32_bf16<<<4096, 256, 0, stream>>>(w_k, wkb, 1048576);
    cvt_f32_bf16<<<4096, 256, 0, stream>>>(w_v, wvb, 1048576);
    cvt_f32_bf16<<<4096, 256, 0, stream>>>(w_o, wob, 1048576);

    dim3 gg(N / 64, M / 128);
    gemm_bf16<<<gg, 256, 0, stream>>>(xb, wqb, nullptr, qB, M, N, K, 1);
    gemm_bf16<<<gg, 256, 0, stream>>>(xb, wkb, nullptr, kB, M, N, K, 1);
    gemm_bf16<<<gg, 256, 0, stream>>>(xb, wvb, nullptr, vT, M, N, K, 2);

    rmsnorm_rope<<<8192, 256, 0, stream>>>(qB, 2048, 65536);
    rmsnorm_rope<<<8192, 256, 0, stream>>>(kB, 2048, 65536);

    dim3 gf(32, 16, 2);
    flash_attn<<<gf, 128, 0, stream>>>(qB, kB, vT, aoB);

    gemm_bf16<<<gg, 256, 0, stream>>>(aoB, wob, (float*)d_out, nullptr, M, N, K, 0);
}
